// FNOBranchNet_38568806318093
// MI455X (gfx1250) — compile-verified, matrix operand order checked by reading, not run on hardware
//
#include <hip/hip_runtime.h>
#include <hip/hip_bf16.h>

typedef __attribute__((ext_vector_type(2)))  float  v2f;
typedef __attribute__((ext_vector_type(8)))  float  v8f;
typedef __attribute__((ext_vector_type(4)))  int    v4i;
typedef __attribute__((ext_vector_type(16))) __bf16 v16bf;

#define WMMA4(a,b,c)    __builtin_amdgcn_wmma_f32_16x16x4_f32(false,(a),false,(b),(short)0,(c),false,false)
#define WMMA32BF(a,b,c) __builtin_amdgcn_wmma_f32_16x16x32_bf16(false,(a),false,(b),(short)0,(c),false,false)

// Problem dims
constexpr int  Bsz  = 16;
constexpr int  Adim = 48;
constexpr int  Gdim = 64;
constexpr int  Cdim = 768;
constexpr int  Wf   = 33;
constexpr int  WP   = 36;            // Wf padded to multiple of 4 (K-tiling)
constexpr int  NBk  = 8;
constexpr long Ntok = (long)Adim * Gdim;        // 3072
constexpr long SLAB = (long)WP * Cdim;          // 27648 floats per (b,a) freq slab
constexpr float LAM = 0.01f;

// ---------------------------------------------------------------------------
// CDNA5 async global->LDS copy (ASYNCcnt path), with safe fallback.
// Builtin parameters are typed: (v4i addrspace(1)*, v4i addrspace(3)*, imm, imm)
// ---------------------------------------------------------------------------
#if __has_builtin(__builtin_amdgcn_global_load_async_to_lds_b128)
#define HAVE_GASYNC 1
#endif

__device__ __forceinline__ void cp_async16(const float* g, float* l) {
#ifdef HAVE_GASYNC
  __builtin_amdgcn_global_load_async_to_lds_b128(
      (__attribute__((address_space(1))) v4i*)(g),
      (__attribute__((address_space(3))) v4i*)(l), 0, 0);
#else
  float4 tmp = *(const float4*)g;
  *(float4*)l = tmp;
#endif
}

__device__ __forceinline__ void wait_async() {
#ifdef HAVE_GASYNC
  asm volatile("s_wait_asynccnt 0x0" ::: "memory");
#endif
}

// ---------------------------------------------------------------------------
// Twiddle tables (ortho norms folded in). Padded rows/cols are zero so the
// WP=36 padding is numerically inert everywhere downstream.
// ---------------------------------------------------------------------------
__global__ void twiddle_kernel(float* RC, float* RS, float* FC, float* FS,
                               float* FSn, float* IRC, float* IRS) {
  int t = blockIdx.x * blockDim.x + threadIdx.x;
  int stride = gridDim.x * blockDim.x;
  const float PI2 = 6.283185307179586f;
  const float s64 = 0.125f;                 // 1/sqrt(64)
  const float s48 = 0.14433756729740643f;   // 1/sqrt(48)
  for (int i = t; i < 48 * 64; i += stride) {
    int w = i / 64, g = i % 64;
    float th = PI2 * (float)(w * g) / 64.f;
    RC[i] = (w < Wf) ?  cosf(th) * s64 : 0.f;
    RS[i] = (w < Wf) ? -sinf(th) * s64 : 0.f;
  }
  for (int i = t; i < 48 * 48; i += stride) {
    int k = i / 48, a = i % 48;
    float th = PI2 * (float)(k * a) / 48.f;
    FC[i]  = cosf(th) * s48;
    FS[i]  = sinf(th) * s48;
    FSn[i] = -FS[i];
  }
  for (int i = t; i < 64 * WP; i += stride) {
    int g = i / WP, w = i % WP;
    float th = PI2 * (float)(g * w) / 64.f;
    float cw = (w == 0 || w == 32) ? 1.f : 2.f;
    bool  v  = (w < Wf);
    IRC[i] = v ?  cw * cosf(th) * s64 : 0.f;
    IRS[i] = v ? -cw * sinf(th) * s64 : 0.f;
  }
}

// ---------------------------------------------------------------------------
// Bias path: out = x @ Wb^T + bb   (M=49152, N=K=768), bf16 WMMA.
// Workgroup = 256 threads (8 waves) computes a 64x64 output tile.
// K sliced by 32; A/B 64x32 slices staged in LDS via async b128 DMA,
// double-buffered so the next slice's DMA overlaps the current WMMAs.
// LDS row stride padded to 36 floats (144B) -> conflict-free fragment reads.
// ---------------------------------------------------------------------------
constexpr int TK   = 32;
constexpr int LDST = 36;

__global__ void bias_gemm_kernel(const float* __restrict__ x,
                                 const float* __restrict__ Wb,
                                 const float* __restrict__ bb,
                                 float* __restrict__ out) {
  __shared__ float sA[2][64 * LDST];
  __shared__ float sB[2][64 * LDST];
  int t = threadIdx.x;
  int wv = t >> 5, lane = t & 31, half = lane >> 4, lr = lane & 15;
  int blkN = blockIdx.x % 12, blkM = blockIdx.x / 12;
  int M0 = blkM * 64, N0 = blkN * 64;
  // 512 b128 chunks per matrix per K-slice; 2 per thread per matrix
  int ch0 = t, ch1 = t + 256;
  int r0 = ch0 >> 3, q0 = (ch0 & 7) * 4;
  int r1 = ch1 >> 3, q1 = (ch1 & 7) * 4;

  auto stage = [&](int buf, int k0) {
    cp_async16(x  + (long)(M0 + r0) * Cdim + k0 + q0, &sA[buf][r0 * LDST + q0]);
    cp_async16(x  + (long)(M0 + r1) * Cdim + k0 + q1, &sA[buf][r1 * LDST + q1]);
    cp_async16(Wb + (long)(N0 + r0) * Cdim + k0 + q0, &sB[buf][r0 * LDST + q0]);
    cp_async16(Wb + (long)(N0 + r1) * Cdim + k0 + q1, &sB[buf][r1 * LDST + q1]);
  };

  int mloc = (wv & 3) * 16;        // 4 m-tiles
  int nloc = (wv >> 2) * 32;       // 2 wave-groups x 2 n-tiles each
  v8f acc0 = {}, acc1 = {};
  int buf = 0;
  stage(0, 0);
  for (int s = 0; s < Cdim / TK; ++s) {
    wait_async();                  // this wave's DMA for `buf` complete
    __syncthreads();               // all waves' DMA complete & visible
    if (s + 1 < Cdim / TK) stage(buf ^ 1, (s + 1) * TK);   // overlap next slice
    v16bf af, bf0, bf1;
#pragma unroll
    for (int e = 0; e < 16; ++e) {
      int v  = e >> 1;
      int kk = 2 * v + (v >= 4 ? 8 : 0) + 8 * half + (e & 1);
      af[e]  = (__bf16)sA[buf][(mloc + lr) * LDST + kk];
      bf0[e] = (__bf16)sB[buf][(nloc + lr) * LDST + kk];
      bf1[e] = (__bf16)sB[buf][(nloc + 16 + lr) * LDST + kk];
    }
    acc0 = WMMA32BF(af, bf0, acc0);
    acc1 = WMMA32BF(af, bf1, acc1);
    __syncthreads();               // all reads of `buf` done before reuse
    buf ^= 1;
  }
  float bias0 = bb[N0 + nloc + lr];
  float bias1 = bb[N0 + nloc + 16 + lr];
#pragma unroll
  for (int j = 0; j < 8; ++j) {
    long m = M0 + mloc + j + 8 * half;
    out[m * Cdim + N0 + nloc + lr]      = acc0[j] + bias0;
    out[m * Cdim + N0 + nloc + 16 + lr] = acc1[j] + bias1;
  }
}

// ---------------------------------------------------------------------------
// Stage 1: rDFT over G per (b,a) slab:  Y[w, c] = DFT64[w, :] @ x[b,a,:,c]
// f32 WMMA 16x16x4; real+imag accumulators share the B fragment.
// ---------------------------------------------------------------------------
__global__ void rdft_g_kernel(const float* __restrict__ x,
                              const float* __restrict__ RC,
                              const float* __restrict__ RS,
                              float* __restrict__ Yr, float* __restrict__ Yi) {
  int wave = blockIdx.x * (blockDim.x >> 5) + (threadIdx.x >> 5);
  int lane = threadIdx.x & 31, half = lane >> 4, lr = lane & 15;
  int nt = wave % 48; wave /= 48;
  int mtile = wave % 3; wave /= 3;
  int a = wave % 48, b = wave / 48;
  int m0 = mtile * 16, n0 = nt * 16;
  const float* xs = x + ((long)b * Ntok + (long)a * Gdim) * Cdim;
  v8f ar = {}, ai = {};
  for (int k0 = 0; k0 < 64; k0 += 4) {
    int row = m0 + lr, kc = k0 + 2 * half;
    v2f ac, as, bf;
    ac.x = RC[row * 64 + kc]; ac.y = RC[row * 64 + kc + 1];
    as.x = RS[row * 64 + kc]; as.y = RS[row * 64 + kc + 1];
    bf.x = xs[(long)kc * Cdim + n0 + lr];
    bf.y = xs[(long)(kc + 1) * Cdim + n0 + lr];
    ar = WMMA4(ac, bf, ar);
    ai = WMMA4(as, bf, ai);
  }
  float* yrs = Yr + ((long)b * 48 + a) * SLAB;
  float* yis = Yi + ((long)b * 48 + a) * SLAB;
#pragma unroll
  for (int j = 0; j < 8; ++j) {
    int m = m0 + j + 8 * half;
    if (m < WP) {
      yrs[(long)m * Cdim + n0 + lr] = ar[j];
      yis[(long)m * Cdim + n0 + lr] = ai[j];
    }
  }
}

// ---------------------------------------------------------------------------
// Complex DFT over A per b:  Out = [FC | T1; T2 | FC] applied to (InR, InI).
// Forward: T1=FS,  T2=FSn.   Inverse: T1=FSn, T2=FS.   (matrices symmetric)
// M=48, K=48, N=WP*C=27648.  4 WMMAs per K-step.
// ---------------------------------------------------------------------------
__global__ void dft_a_kernel(const float* __restrict__ Ir, const float* __restrict__ Ii,
                             const float* __restrict__ FC, const float* __restrict__ T1,
                             const float* __restrict__ T2,
                             float* __restrict__ OutR, float* __restrict__ OutI) {
  int wave = blockIdx.x * (blockDim.x >> 5) + (threadIdx.x >> 5);
  int lane = threadIdx.x & 31, half = lane >> 4, lr = lane & 15;
  int nt = wave % 1728; wave /= 1728;
  int mtile = wave % 3;  int b = wave / 3;
  int m0 = mtile * 16, n0 = nt * 16;
  const float* pr = Ir + (long)b * 48 * SLAB;
  const float* pi = Ii + (long)b * 48 * SLAB;
  v8f zr = {}, zi = {};
  for (int k0 = 0; k0 < 48; k0 += 4) {
    int row = m0 + lr, kc = k0 + 2 * half;
    v2f fc, t1, t2, br, bi;
    fc.x = FC[row * 48 + kc]; fc.y = FC[row * 48 + kc + 1];
    t1.x = T1[row * 48 + kc]; t1.y = T1[row * 48 + kc + 1];
    t2.x = T2[row * 48 + kc]; t2.y = T2[row * 48 + kc + 1];
    br.x = pr[(long)kc * SLAB + n0 + lr]; br.y = pr[(long)(kc + 1) * SLAB + n0 + lr];
    bi.x = pi[(long)kc * SLAB + n0 + lr]; bi.y = pi[(long)(kc + 1) * SLAB + n0 + lr];
    zr = WMMA4(fc, br, zr);  zr = WMMA4(t1, bi, zr);
    zi = WMMA4(fc, bi, zi);  zi = WMMA4(t2, br, zi);
  }
#pragma unroll
  for (int j = 0; j < 8; ++j) {
    int m = m0 + j + 8 * half;
    OutR[((long)b * 48 + m) * SLAB + n0 + lr] = zr[j];
    OutI[((long)b * 48 + m) * SLAB + n0 + lr] = zi[j];
  }
}

// ---------------------------------------------------------------------------
// Block-diagonal complex MLP. One workgroup = 6 waves = one 16-position tile
// of one block nb. Layer-1 H (16x96, r+i) staged in LDS; layer-2 fused with
// bias + softshrink. f32 WMMA 16x16x4, K=96.
// ---------------------------------------------------------------------------
__global__ void mlp_kernel(const float* __restrict__ Xr, const float* __restrict__ Xi,
                           const float* __restrict__ w1, const float* __restrict__ b1,
                           const float* __restrict__ w2, const float* __restrict__ b2,
                           float* __restrict__ OutR, float* __restrict__ OutI) {
  __shared__ float sHr[16][96];
  __shared__ float sHi[16][96];
  int wv = threadIdx.x >> 5;                     // 0..5 -> n-column tile of 96
  int lane = threadIdx.x & 31, half = lane >> 4, lr = lane & 15;
  int nb   = blockIdx.x & 7;
  int tile = blockIdx.x >> 3;                    // 0..1727 position tiles
  long rowbase = ((long)tile * 16 + lr) * Cdim + (long)nb * 96;
  const float* W10 = w1 + ((long)0 * NBk + nb) * 96 * 96;
  const float* W11 = w1 + ((long)1 * NBk + nb) * 96 * 96;
  const float* W20 = w2 + ((long)0 * NBk + nb) * 96 * 96;
  const float* W21 = w2 + ((long)1 * NBk + nb) * 96 * 96;
  int n0 = wv * 16;

  v8f hr = {}, hi = {};
  for (int k0 = 0; k0 < 96; k0 += 4) {
    int kc = k0 + 2 * half;
    v2f axr, axi, b0, bf1, bn1;
    axr.x = Xr[rowbase + kc]; axr.y = Xr[rowbase + kc + 1];
    axi.x = Xi[rowbase + kc]; axi.y = Xi[rowbase + kc + 1];
    b0.x  = W10[kc * 96 + n0 + lr];  b0.y  = W10[(kc + 1) * 96 + n0 + lr];
    bf1.x = W11[kc * 96 + n0 + lr];  bf1.y = W11[(kc + 1) * 96 + n0 + lr];
    bn1.x = -bf1.x;                  bn1.y = -bf1.y;
    hr = WMMA4(axr, b0,  hr);  hr = WMMA4(axi, bn1, hr);   // Xr@W10 - Xi@W11
    hi = WMMA4(axr, bf1, hi);  hi = WMMA4(axi, b0,  hi);   // Xr@W11 + Xi@W10
  }
  float biasR = b1[(0 * NBk + nb) * 96 + n0 + lr];
  float biasI = b1[(1 * NBk + nb) * 96 + n0 + lr];
#pragma unroll
  for (int j = 0; j < 8; ++j) {
    int m = j + 8 * half;
    sHr[m][n0 + lr] = fmaxf(hr[j] + biasR, 0.f);
    sHi[m][n0 + lr] = fmaxf(hi[j] + biasI, 0.f);
  }
  __syncthreads();

  v8f orr = {}, oii = {};
  for (int k0 = 0; k0 < 96; k0 += 4) {
    int kc = k0 + 2 * half;
    v2f ahr, ahi, c0, c1, c1n;
    ahr.x = sHr[lr][kc]; ahr.y = sHr[lr][kc + 1];
    ahi.x = sHi[lr][kc]; ahi.y = sHi[lr][kc + 1];
    c0.x  = W20[kc * 96 + n0 + lr]; c0.y  = W20[(kc + 1) * 96 + n0 + lr];
    c1.x  = W21[kc * 96 + n0 + lr]; c1.y  = W21[(kc + 1) * 96 + n0 + lr];
    c1n.x = -c1.x;                  c1n.y = -c1.y;
    orr = WMMA4(ahr, c0,  orr);  orr = WMMA4(ahi, c1n, orr);
    oii = WMMA4(ahr, c1,  oii);  oii = WMMA4(ahi, c0,  oii);
  }
  float b2R = b2[(0 * NBk + nb) * 96 + n0 + lr];
  float b2I = b2[(1 * NBk + nb) * 96 + n0 + lr];
#pragma unroll
  for (int j = 0; j < 8; ++j) {
    int m = j + 8 * half;
    long orow = ((long)tile * 16 + m) * Cdim + (long)nb * 96 + n0 + lr;
    float vr = orr[j] + b2R, vi = oii[j] + b2I;
    vr = vr > LAM ? vr - LAM : (vr < -LAM ? vr + LAM : 0.f);
    vi = vi > LAM ? vi - LAM : (vi < -LAM ? vi + LAM : 0.f);
    OutR[orow] = vr;  OutI[orow] = vi;
  }
}

// ---------------------------------------------------------------------------
// Inverse rDFT over G per (b,a) slab, accumulating onto bias already in d_out:
// out[g, c] += IRC[g, :] @ Zr[:, c] + IRS[g, :] @ Zi[:, c]      (K=WP=36)
// ---------------------------------------------------------------------------
__global__ void irdft_kernel(const float* __restrict__ Zr, const float* __restrict__ Zi,
                             const float* __restrict__ IRC, const float* __restrict__ IRS,
                             float* __restrict__ out) {
  int wave = blockIdx.x * (blockDim.x >> 5) + (threadIdx.x >> 5);
  int lane = threadIdx.x & 31, half = lane >> 4, lr = lane & 15;
  int nt = wave % 48; wave /= 48;
  int mtile = wave % 4; wave /= 4;
  int a = wave % 48, b = wave / 48;
  int m0 = mtile * 16, n0 = nt * 16;
  const float* zr = Zr + ((long)b * 48 + a) * SLAB;
  const float* zi = Zi + ((long)b * 48 + a) * SLAB;
  v8f acc = {};
  for (int k0 = 0; k0 < WP; k0 += 4) {
    int row = m0 + lr, kc = k0 + 2 * half;
    v2f ac, as, br, bi;
    ac.x = IRC[row * WP + kc]; ac.y = IRC[row * WP + kc + 1];
    as.x = IRS[row * WP + kc]; as.y = IRS[row * WP + kc + 1];
    br.x = zr[(long)kc * Cdim + n0 + lr]; br.y = zr[(long)(kc + 1) * Cdim + n0 + lr];
    bi.x = zi[(long)kc * Cdim + n0 + lr]; bi.y = zi[(long)(kc + 1) * Cdim + n0 + lr];
    acc = WMMA4(ac, br, acc);
    acc = WMMA4(as, bi, acc);
  }
#pragma unroll
  for (int j = 0; j < 8; ++j) {
    int g = m0 + j + 8 * half;
    long idx = ((long)b * Ntok + (long)a * Gdim + g) * Cdim + n0 + lr;
    out[idx] += acc[j];
  }
}

// ---------------------------------------------------------------------------
extern "C" void kernel_launch(void* const* d_in, const int* in_sizes, int n_in,
                              void* d_out, int out_size, void* d_ws, size_t ws_size,
                              hipStream_t stream) {
  const float* x  = (const float*)d_in[0];
  const float* w1 = (const float*)d_in[1];
  const float* b1 = (const float*)d_in[2];
  const float* w2 = (const float*)d_in[3];
  const float* b2 = (const float*)d_in[4];
  const float* Wb = (const float*)d_in[5];
  const float* bb = (const float*)d_in[6];
  float* out = (float*)d_out;
  float* ws  = (float*)d_ws;

  size_t off = 0;
  float* RC  = ws + off; off += 48 * 64;
  float* RS  = ws + off; off += 48 * 64;
  float* FC  = ws + off; off += 48 * 48;
  float* FS  = ws + off; off += 48 * 48;
  float* FSn = ws + off; off += 48 * 48;
  float* IRC = ws + off; off += 64 * WP;
  float* IRS = ws + off; off += 64 * WP;
  off = (off + 255) & ~(size_t)255;
  const size_t NBUF = (size_t)Bsz * 48 * WP * Cdim;   // 21,233,664 floats
  float* P0r = ws + off; off += NBUF;
  float* P0i = ws + off; off += NBUF;
  float* P1r = ws + off; off += NBUF;
  float* P1i = ws + off; off += NBUF;

  twiddle_kernel<<<30, 256, 0, stream>>>(RC, RS, FC, FS, FSn, IRC, IRS);

  // bias GEMM: (49152/64) x (768/64) = 768 x 12 = 9216 workgroups
  bias_gemm_kernel<<<768 * 12, 256, 0, stream>>>(x, Wb, bb, out);

  // forward rDFT over G: 768 slabs x 3 m-tiles x 48 n-tiles = 110592 waves
  rdft_g_kernel<<<110592 / 8, 256, 0, stream>>>(x, RC, RS, P0r, P0i);

  // forward DFT over A: 16 x 3 x 1728 = 82944 waves
  dft_a_kernel<<<82944 / 8, 256, 0, stream>>>(P0r, P0i, FC, FS, FSn, P1r, P1i);

  // block-diagonal complex MLP: 1728 tiles x 8 blocks, 6 waves each
  mlp_kernel<<<1728 * 8, 192, 0, stream>>>(P1r, P1i, w1, b1, w2, b2, P0r, P0i);

  // inverse DFT over A (swap sin tables)
  dft_a_kernel<<<82944 / 8, 256, 0, stream>>>(P0r, P0i, FC, FSn, FS, P1r, P1i);

  // inverse rDFT over G, accumulate into d_out: 768 x 4 x 48 = 147456 waves
  irdft_kernel<<<147456 / 8, 256, 0, stream>>>(P1r, P1i, IRC, IRS, out);
}
